// MambaLayer_60550448939203
// MI455X (gfx1250) — matrix operand order repaired and checked by reading, not compiled, not run on hardware
//
#include <hip/hip_runtime.h>
#include <hip/hip_bf16.h>
#include <cmath>

typedef __attribute__((ext_vector_type(16))) _Float16 v16h;
typedef __attribute__((ext_vector_type(8)))  _Float16 v8h;
typedef __attribute__((ext_vector_type(4)))  _Float16 v4h;
typedef __attribute__((ext_vector_type(8)))  float    v8f;
typedef __attribute__((ext_vector_type(4)))  float    v4f;

#define WMMA_F32_F16(a, b, c) \
  __builtin_amdgcn_wmma_f32_16x16x32_f16(false, (a), false, (b), (short)0, (c), false, false)

static __device__ __forceinline__ v16h hcat(v8h lo, v8h hi) {
  return __builtin_shufflevector(lo, hi, 0, 1, 2, 3, 4, 5, 6, 7,
                                 8, 9, 10, 11, 12, 13, 14, 15);
}

static constexpr int Bb = 8, Ll = 2048, Hh = 256, Ss = 64, Ee = 512;
static constexpr int ML = Bb * Ll; // 16384 rows (B*L)

// ---------------------------------------------------------------------------
// GEMM with bias: C[M,N] = A[M,K]*B[K,N] + bias[N]. f32 in/out, f16 WMMA.
// Block = 128 threads (4 waves), 64x64 tile; wave computes 16(M)x64(N).
// A staged row-major f16 (fragment rows contiguous per lane); B staged
// directly in WMMA-B fragment layout so reads are contiguous 32B per lane.
// ---------------------------------------------------------------------------
__global__ __launch_bounds__(128)
void gemm_bias_wmma(const float* __restrict__ A, const float* __restrict__ Bm,
                    const float* __restrict__ bias, float* __restrict__ C,
                    int M, int N, int K) {
  __shared__ _Float16 As[64][40];        // 64 rows x 32 k (+pad, 16B-aligned rows)
  __shared__ _Float16 BsF[4][32][16];    // [ntile][lane][h]  (fragment layout)
  const int tid  = threadIdx.x;
  const int wave = tid >> 5, lane = tid & 31;
  const int m15  = lane & 15, hi = lane >> 4;
  const int n0 = blockIdx.x * 64;
  const int m0 = blockIdx.y * 64;
  // A-loader mapping: 4-float chunks along K
  const int ar = tid >> 3;               // 0..15
  const int ac = (tid & 7) * 4;          // 0..28 step 4
  // B-loader mapping: column bc, side brg, 4-row groups
  const int bc   = tid & 63;
  const int brg  = tid >> 6;             // 0..1 (k side)
  const int bnt  = bc >> 4, bn = bc & 15;
  const int blane = brg * 16 + bn;

  v8f acc[4] = {};
  for (int kk = 0; kk < K; kk += 32) {
    #pragma unroll
    for (int p = 0; p < 4; ++p) {
      int row = p * 16 + ar;
      v4f f = *(const v4f*)&A[(size_t)(m0 + row) * K + (kk + ac)];
      v4h h;
      #pragma unroll
      for (int q = 0; q < 4; ++q) h[q] = (_Float16)f[q];
      *(v4h*)&As[row][ac] = h;
    }
    #pragma unroll
    for (int g = 0; g < 4; ++g) {
      int r0 = brg * 16 + g * 4;
      v4h h;
      #pragma unroll
      for (int q = 0; q < 4; ++q)
        h[q] = (_Float16)Bm[(size_t)(kk + r0 + q) * N + (n0 + bc)];
      *(v4h*)&BsF[bnt][blane][g * 4] = h;
    }
    __syncthreads();

    const _Float16* arow = &As[wave * 16 + m15][0];
    v16h af = hcat(*(const v8h*)&arow[hi * 16], *(const v8h*)&arow[hi * 16 + 8]);
    #pragma unroll
    for (int nt = 0; nt < 4; ++nt) {
      v16h bf = *(const v16h*)&BsF[nt][lane][0];
      acc[nt] = WMMA_F32_F16(af, bf, acc[nt]);
    }
    __syncthreads();
  }
  #pragma unroll
  for (int nt = 0; nt < 4; ++nt) {
    int col = n0 + nt * 16 + m15;
    float bcol = bias[col];
    #pragma unroll
    for (int v = 0; v < 8; ++v) {
      int row = m0 + wave * 16 + v + (hi << 3);
      C[(size_t)row * N + col] = acc[nt][v] + bcol;
    }
  }
}

// ---------------------------------------------------------------------------
// Depthwise conv1d (k=3, pad=1 along L) + SiLU on the x-half of xz.
// ---------------------------------------------------------------------------
__global__ __launch_bounds__(256)
void conv_silu_kernel(const float* __restrict__ xz, const float* __restrict__ conv_w,
                      const float* __restrict__ conv_b, float* __restrict__ xc) {
  int idx = blockIdx.x * 256 + threadIdx.x;  // over B*L*E
  if (idx >= Bb * Ll * Ee) return;
  int e = idx % Ee;
  int l = (idx / Ee) % Ll;
  int b = idx / (Ee * Ll);
  size_t base = (size_t)b * Ll * (2 * Ee);
  float xm = (l > 0)      ? xz[base + (size_t)(l - 1) * (2 * Ee) + e] : 0.f;
  float x0 =                xz[base + (size_t)l       * (2 * Ee) + e];
  float xp = (l < Ll - 1) ? xz[base + (size_t)(l + 1) * (2 * Ee) + e] : 0.f;
  float v = xm * conv_w[e * 3 + 0] + x0 * conv_w[e * 3 + 1] + xp * conv_w[e * 3 + 2]
          + conv_b[e];
  xc[idx] = v / (1.f + __expf(-v));  // silu
}

// ---------------------------------------------------------------------------
// Per-row scan inputs: delta = softplus(ssm[:,0:64]); it = delta*Bm*sum_e(x);
// C = ssm[:,128:192]. One block per (b,t) row.
// ---------------------------------------------------------------------------
__global__ __launch_bounds__(256)
void ssm_prep_kernel(const float* __restrict__ ssm, const float* __restrict__ xc,
                     float* __restrict__ dOut, float* __restrict__ itOut,
                     float* __restrict__ cOut) {
  __shared__ float red[256];
  int row = blockIdx.x, tid = threadIdx.x;
  const float* xr = xc + (size_t)row * Ee;
  red[tid] = xr[tid] + xr[tid + 256];
  __syncthreads();
  for (int off = 128; off > 0; off >>= 1) {
    if (tid < off) red[tid] += red[tid + off];
    __syncthreads();
  }
  float xsum = red[0];
  if (tid < Ss) {
    const float* sr = ssm + (size_t)row * (3 * Ss);
    float v = sr[tid];
    float d = (v > 20.f) ? v : log1pf(__expf(v));  // softplus
    dOut[(size_t)row * Ss + tid]  = d;
    itOut[(size_t)row * Ss + tid] = d * sr[Ss + tid] * xsum;
    cOut[(size_t)row * Ss + tid]  = sr[2 * Ss + tid];
  }
}

// ---------------------------------------------------------------------------
// Sequential SSM scan. One block per (batch, E-half): 16 blocks, 256 threads
// (8 waves). State h[64][256-cols] lives in LDS *in WMMA-B fragment layout*
// (f16): hF[etile][ktile][lane][h], so B-operand reads are contiguous 32B per
// lane and D-tile writes are contiguous 16B per lane. A_d rebuilt each step.
// ---------------------------------------------------------------------------
__global__ __launch_bounds__(256)
void scan_wmma_kernel(const float* __restrict__ dIn, const float* __restrict__ itIn,
                      const float* __restrict__ cIn, const float* __restrict__ xc,
                      const float* __restrict__ A_log, const float* __restrict__ Dp,
                      float* __restrict__ outputs) {
  __shared__ _Float16 hF[16][2][32][16];  // 32 KB state, fragment layout
  __shared__ _Float16 Ad[64][72];         // A_d row-major (rows 16B-aligned)
  __shared__ float dS[64], cS[64], iS[64], aS[64];
  const int tid  = threadIdx.x;
  const int b    = blockIdx.x >> 1;
  const int ec   = (blockIdx.x & 1) * 256;  // E-chunk base
  const int wave = tid >> 5, lane = tid & 31;
  const int m15  = lane & 15, hi = lane >> 4;
  const int adi  = tid >> 2;              // A_d row owned by this thread
  const int adj  = (tid & 3) * 16;        // 16 consecutive cols

  if (tid < 64) {
    float a = fminf(fmaxf(A_log[tid], -5.f), 5.f);
    aS[tid] = -__expf(a);
  }
  {
    v8h z = {};
    v8h* hp = (v8h*)&hF[0][0][0][0];
    #pragma unroll
    for (int q = 0; q < 8; ++q) hp[q * 256 + tid] = z;  // 2048 v8h total
  }
  __syncthreads();

  for (int t = 0; t < Ll; ++t) {
    size_t rbase = ((size_t)b * Ll + t) * Ss;
    if (tid < 64) {
      dS[tid] = dIn[rbase + tid];
      cS[tid] = cIn[rbase + tid];
      iS[tid] = itIn[rbase + tid];
      if (t + 1 < Ll) {  // prefetch next step's rows
        __builtin_prefetch(dIn + rbase + Ss, 0, 0);
        __builtin_prefetch(cIn + rbase + Ss, 0, 0);
        __builtin_prefetch(itIn + rbase + Ss, 0, 0);
      }
    }
    __syncthreads();
    // build A_d: each thread owns 16 consecutive cols of one row
    {
      float di = dS[adi];
      v8h h0, h1;
      #pragma unroll
      for (int q = 0; q < 8; ++q) {
        float v0 = fminf(fmaxf(di * aS[adj + q], -10.f), 0.f);
        float v1 = fminf(fmaxf(di * aS[adj + 8 + q], -10.f), 0.f);
        h0[q] = (_Float16)__expf(v0);
        h1[q] = (_Float16)__expf(v1);
      }
      *(v8h*)&Ad[adi][adj]     = h0;
      *(v8h*)&Ad[adi][adj + 8] = h1;
    }
    __syncthreads();

    // A fragments (per-lane contiguous 16B chunks)
    v16h a[4][2];
    #pragma unroll
    for (int i = 0; i < 4; ++i) {
      const _Float16* arow = &Ad[i * 16 + m15][0];
      #pragma unroll
      for (int j = 0; j < 2; ++j)
        a[i][j] = hcat(*(const v8h*)&arow[j * 32 + hi * 16],
                       *(const v8h*)&arow[j * 32 + hi * 16 + 8]);
    }
    // C fragments (input term) and y weights: rows i*16 + 8*hi + (0..7) are
    // contiguous, 32B-aligned -> vector LDS loads (2x ds_load_b128 each).
    v8f cfrag[4], cwt[4];
    #pragma unroll
    for (int i = 0; i < 4; ++i) {
      cfrag[i] = *(const v8f*)&iS[i * 16 + (hi << 3)];
      cwt[i]   = *(const v8f*)&cS[i * 16 + (hi << 3)];
    }

    #pragma unroll
    for (int et = 0; et < 2; ++et) {
      int etile = wave * 2 + et;
      v16h b0 = *(const v16h*)&hF[etile][0][lane][0];
      v16h b1 = *(const v16h*)&hF[etile][1][lane][0];
      float yacc = 0.f;
      #pragma unroll
      for (int i = 0; i < 4; ++i) {
        v8f t0 = WMMA_F32_F16(a[i][0], b0, cfrag[i]);
        v8f d  = WMMA_F32_F16(a[i][1], b1, t0);
        v8h hh;
        #pragma unroll
        for (int v = 0; v < 8; ++v) {
          float hv = fminf(fmaxf(d[v], -100.f), 100.f);
          hh[v] = (_Float16)hv;
          yacc += hv * cwt[i][v];
        }
        // D(row srow = i*16 + v + 8*hi, col m15) -> fragment slot:
        // ktile = i>>1, dest lane = (i&1)*16 + m15, h = v + 8*hi
        *(v8h*)&hF[etile][i >> 1][(i & 1) * 16 + m15][hi * 8] = hh;
      }
      yacc += __shfl_xor(yacc, 16, 32);  // combine the two m-halves
      if (hi == 0) {
        int ge = ec + etile * 16 + m15;
        size_t oidx = ((size_t)b * Ll + t) * Ee + ge;
        outputs[oidx] = yacc + Dp[ge] * xc[oidx];
      }
    }
    __syncthreads();
  }
}

// ---------------------------------------------------------------------------
// LayerNorm over E + SiLU(z) gating. One block per (b,t) row.
// ---------------------------------------------------------------------------
__global__ __launch_bounds__(256)
void ln_gate_kernel(const float* __restrict__ outputs, const float* __restrict__ xz,
                    const float* __restrict__ ln_g, const float* __restrict__ ln_b,
                    float* __restrict__ g) {
  __shared__ float r1[256], r2[256];
  int row = blockIdx.x, tid = threadIdx.x;
  const float* orow = outputs + (size_t)row * Ee;
  float a = orow[tid], bq = orow[tid + 256];
  r1[tid] = a + bq;
  r2[tid] = a * a + bq * bq;
  __syncthreads();
  for (int off = 128; off > 0; off >>= 1) {
    if (tid < off) { r1[tid] += r1[tid + off]; r2[tid] += r2[tid + off]; }
    __syncthreads();
  }
  float mu  = r1[0] * (1.f / Ee);
  float var = r2[0] * (1.f / Ee) - mu * mu;
  float inv = rsqrtf(var + 1e-5f);
  const float* zrow = xz + (size_t)row * (2 * Ee) + Ee;
  for (int e = tid; e < Ee; e += 256) {
    float y = (orow[e] - mu) * inv * ln_g[e] + ln_b[e];
    float z = zrow[e];
    g[(size_t)row * Ee + e] = y * (z / (1.f + __expf(-z)));
  }
}

// ---------------------------------------------------------------------------
extern "C" void kernel_launch(void* const* d_in, const int* in_sizes, int n_in,
                              void* d_out, int out_size, void* d_ws, size_t ws_size,
                              hipStream_t stream) {
  const float* u      = (const float*)d_in[0];
  const float* W_in   = (const float*)d_in[1];
  const float* b_in   = (const float*)d_in[2];
  const float* conv_w = (const float*)d_in[3];
  const float* conv_b = (const float*)d_in[4];
  const float* W_x    = (const float*)d_in[5];
  const float* b_x    = (const float*)d_in[6];
  const float* A_log  = (const float*)d_in[7];
  const float* Dp     = (const float*)d_in[8];
  const float* W_out  = (const float*)d_in[9];
  const float* b_out  = (const float*)d_in[10];
  const float* ln_g   = (const float*)d_in[11];
  const float* ln_b   = (const float*)d_in[12];
  float* out = (float*)d_out;

  float* ws = (float*)d_ws;
  size_t off = 0;
  float* xz   = ws + off; off += (size_t)ML * 2 * Ee;  // [16384,1024]
  float* xc   = ws + off; off += (size_t)ML * Ee;      // post-conv silu x
  float* ssm  = ws + off; off += (size_t)ML * 3 * Ss;  // delta|B|C raw
  float* dd   = ws + off; off += (size_t)ML * Ss;
  float* it   = ws + off; off += (size_t)ML * Ss;
  float* cc   = ws + off; off += (size_t)ML * Ss;
  float* outs = ws + off; off += (size_t)ML * Ee;
  float* gb   = ws + off; off += (size_t)ML * Ee;

  // 1) xz = u @ W_in + b_in           [16384 x 1024 x 256]
  gemm_bias_wmma<<<dim3((2 * Ee) / 64, ML / 64), 128, 0, stream>>>(
      u, W_in, b_in, xz, ML, 2 * Ee, Hh);
  // 2) depthwise conv + silu
  conv_silu_kernel<<<(ML * Ee) / 256, 256, 0, stream>>>(xz, conv_w, conv_b, xc);
  // 3) ssm = xc @ W_x + b_x           [16384 x 192 x 512]
  gemm_bias_wmma<<<dim3((3 * Ss) / 64, ML / 64), 128, 0, stream>>>(
      xc, W_x, b_x, ssm, ML, 3 * Ss, Ee);
  // 4) per-row scan inputs
  ssm_prep_kernel<<<ML, 256, 0, stream>>>(ssm, xc, dd, it, cc);
  // 5) sequential scan (WMMA recurrence), 16 blocks = 8 batches x 2 E-halves
  scan_wmma_kernel<<<16, 256, 0, stream>>>(dd, it, cc, xc, A_log, Dp, outs);
  // 6) layernorm + silu(z) gate
  ln_gate_kernel<<<ML, 256, 0, stream>>>(outs, xz, ln_g, ln_b, gb);
  // 7) out = gb @ W_out + b_out       [16384 x 256 x 512]
  gemm_bias_wmma<<<dim3(Hh / 64, ML / 64), 128, 0, stream>>>(
      gb, W_out, b_out, out, ML, Hh, Ee);
}
